// LIF_hh_neuron_14620068675619
// MI455X (gfx1250) — compile-verified, matrix operand order checked by reading, not run on hardware
//
#include <hip/hip_runtime.h>
#include <stdint.h>

#define THRESH 0.8f
#define DECAY  0.2f

#define BATCH 2048
#define TSTEP 15
#define TPAD  16
#define INDIM 512
#define CDIM  1024
#define NB 4          // batches per workgroup (4 M-tiles of 16 padded timesteps)
#define NC 64         // channels per workgroup (4 N-tiles)
#define KSTEP 32
#define ROWB 80       // LDS row: 32 bf16 (64B) + 16B pad, keeps 16B alignment
#define TILEB ((NB * TPAD + 3 * NC) * ROWB)   // 20480 B per buffer

#define XPAD_ELEMS ((size_t)BATCH * TPAD * INDIM)       // 16,777,216 bf16
#define WBF_ELEMS  ((size_t)3 * CDIM * INDIM)           //  1,572,864 bf16
#define WS_NEED    (2 * (XPAD_ELEMS + WBF_ELEMS))       // 36,700,160 B

typedef __attribute__((ext_vector_type(16))) __bf16       v16bf;
typedef __attribute__((ext_vector_type(8)))  float        v8f;
typedef __attribute__((ext_vector_type(4)))  float        vf4;
typedef __attribute__((ext_vector_type(4)))  int          v4i;
typedef __attribute__((ext_vector_type(4)))  unsigned int vu4;
typedef __attribute__((ext_vector_type(2)))  unsigned int vu2;

union Frag {
  v16bf v;
  vu4   q[2];
};

__device__ __forceinline__ unsigned int pack2_bf16(float a, float b) {
  unsigned int ua = __float_as_uint(a);
  unsigned int ub = __float_as_uint(b);
  ua = (ua + 0x7FFFu + ((ua >> 16) & 1u)) >> 16;   // round-to-nearest-even
  ub = (ub + 0x7FFFu + ((ub >> 16) & 1u)) >> 16;
  return (ua & 0xFFFFu) | (ub << 16);
}

// ---- CDNA5 async global->LDS copy (ASYNCcnt) with safe fallbacks ----------
__device__ __forceinline__ void async_cp16(const void* g, void* l) {
#if __has_builtin(__builtin_amdgcn_global_load_async_to_lds_b128)
  __builtin_amdgcn_global_load_async_to_lds_b128(
      (__attribute__((address_space(1))) v4i*)g,
      (__attribute__((address_space(3))) v4i*)l, 0, 0);
#else
  *(vu4*)l = *(const vu4*)g;
#endif
}

template <int N>
__device__ __forceinline__ void wait_async() {
#if __has_builtin(__builtin_amdgcn_global_load_async_to_lds_b128)
#if __has_builtin(__builtin_amdgcn_s_wait_asynccnt)
  __builtin_amdgcn_s_wait_asynccnt(N);
#else
  asm volatile("s_wait_asynccnt %0" :: "i"(N) : "memory");
#endif
#endif
}

// ---- shared LIF scan (runs directly on WMMA accumulators) -----------------
// C tile (m,n) = (t, channel): vgpr j holds t = j + 8*half, lane%16 = channel
__device__ __forceinline__ void lif_scan_store(
    const v8f (&acc)[3][2], int b, int cBase, int half, int nloc,
    const float* __restrict__ b1, const float* __restrict__ b2,
    const float* __restrict__ b3, float wl0, float wl1, float wl2, float blv,
    float* __restrict__ out)
{
  #pragma unroll
  for (int i = 0; i < 2; ++i) {
    const int c = cBase + i * 16 + nloc;
    const float b1c = b1[c], b2c = b2[c], b3c = b3[c];
    float m0 = 0.f, m1 = 0.f, m2 = 0.f, m3 = 0.f;
    float s0 = 0.f, s1 = 0.f, s2 = 0.f, s3 = 0.f;

    if (half == 0) {
      #pragma unroll
      for (int j = 0; j < 8; ++j) {             // t = 0..7
        const float inner = m0 * wl0 + m1 * wl1 + m2 * wl2 + blv;
        m0 = m0 * DECAY * (1.f - s0) + acc[0][i][j] + b1c;
        m1 = m1 * DECAY * (1.f - s1) + acc[1][i][j] + b2c;
        m2 = m2 * DECAY * (1.f - s2) + acc[2][i][j] + b3c;
        m3 = m3 * DECAY * (1.f - s3) + inner;
        s0 = m0 > THRESH ? 1.f : 0.f;
        s1 = m1 > THRESH ? 1.f : 0.f;
        s2 = m2 > THRESH ? 1.f : 0.f;
        s3 = m3 > THRESH ? 1.f : 0.f;
        vf4 o; o.x = s0; o.y = s1; o.z = s2; o.w = s3;
        __builtin_nontemporal_store(
            o, (vf4*)(out + ((size_t)(b * TSTEP + j) * CDIM + c) * 4));
      }
    }
    // hand (mem, spike) state from lane c to lane c+16
    m0 = __shfl(m0, nloc, 32); m1 = __shfl(m1, nloc, 32);
    m2 = __shfl(m2, nloc, 32); m3 = __shfl(m3, nloc, 32);
    s0 = __shfl(s0, nloc, 32); s1 = __shfl(s1, nloc, 32);
    s2 = __shfl(s2, nloc, 32); s3 = __shfl(s3, nloc, 32);

    if (half == 1) {
      #pragma unroll
      for (int j = 0; j < 7; ++j) {             // t = 8..14 (t=15 is padding)
        const float inner = m0 * wl0 + m1 * wl1 + m2 * wl2 + blv;
        m0 = m0 * DECAY * (1.f - s0) + acc[0][i][j] + b1c;
        m1 = m1 * DECAY * (1.f - s1) + acc[1][i][j] + b2c;
        m2 = m2 * DECAY * (1.f - s2) + acc[2][i][j] + b3c;
        m3 = m3 * DECAY * (1.f - s3) + inner;
        s0 = m0 > THRESH ? 1.f : 0.f;
        s1 = m1 > THRESH ? 1.f : 0.f;
        s2 = m2 > THRESH ? 1.f : 0.f;
        s3 = m3 > THRESH ? 1.f : 0.f;
        vf4 o; o.x = s0; o.y = s1; o.z = s2; o.w = s3;
        __builtin_nontemporal_store(
            o, (vf4*)(out + ((size_t)(b * TSTEP + 8 + j) * CDIM + c) * 4));
      }
    }
  }
}

// ---- pre-conversion kernels ----------------------------------------------
// xpad[b][t][k] bf16, t padded to 16 with zeros
__global__ __launch_bounds__(256)
void cvt_x_kernel(const float* __restrict__ x, __bf16* __restrict__ xpad) {
  const unsigned g = blockIdx.x * 256 + threadIdx.x;   // one float4 group
  const int part = g & 127;
  const int t    = (g >> 7) & 15;
  const int b    = g >> 11;
  vf4 v = (vf4)(0.0f);
  if (t < TSTEP)
    v = *(const vf4*)(x + (size_t)(b * TSTEP + t) * INDIM + part * 4);
  vu2 pk;
  pk.x = pack2_bf16(v.x, v.y);
  pk.y = pack2_bf16(v.z, v.w);
  *(vu2*)(xpad + (size_t)(b * TPAD + t) * INDIM + part * 4) = pk;
}

__global__ __launch_bounds__(256)
void cvt_w_kernel(const float* __restrict__ W, __bf16* __restrict__ dst) {
  const unsigned g = blockIdx.x * 256 + threadIdx.x;   // one float4 group
  vf4 v = *(const vf4*)(W + (size_t)g * 4);
  vu2 pk;
  pk.x = pack2_bf16(v.x, v.y);
  pk.y = pack2_bf16(v.z, v.w);
  *(vu2*)(dst + (size_t)g * 4) = pk;
}

// ---- stage one K-slab of A+B tiles into an LDS buffer (4 async instrs/wave)
__device__ __forceinline__ void stage_tiles(
    const __bf16* __restrict__ xpad, const __bf16* __restrict__ wbf,
    unsigned char* buf, int bb, int cb, int k0, int tid)
{
  {
    const int row = tid >> 2, chunk = tid & 3;        // 64 rows x 4 x 16B
    const int b = bb * NB + (row >> 4), t = row & 15;
    const __bf16* g = xpad + ((size_t)(b * TPAD + t) * INDIM + k0 + chunk * 8);
    async_cp16(g, buf + row * ROWB + chunk * 16);
  }
  #pragma unroll
  for (int it = 0; it < 3; ++it) {                    // 192 rows x 4 x 16B
    const int s = tid + it * 256;
    const int row = s >> 2, chunk = s & 3;
    const int w = row >> 6, n = row & 63;
    const __bf16* g =
        wbf + ((size_t)(w * CDIM + cb * NC + n) * INDIM + k0 + chunk * 8);
    async_cp16(g, buf + (NB * TPAD + row) * ROWB + chunk * 16);
  }
}

// ---- main fused kernel: bf16 inputs, async double-buffered staging --------
__global__ __launch_bounds__(256)
void lif_hh_fused_bf16_kernel(const __bf16* __restrict__ xpad,
                              const __bf16* __restrict__ wbf,
                              const float* __restrict__ b1,
                              const float* __restrict__ b2,
                              const float* __restrict__ b3,
                              const float* __restrict__ Wl,
                              const float* __restrict__ bl,
                              float* __restrict__ out)
{
  __shared__ __align__(16) unsigned char smem[2][TILEB];   // 40 KB

  const int tid    = threadIdx.x;
  const int lane   = tid & 31;
  const int wid    = tid >> 5;
  const int wave_m = wid >> 1;
  const int wave_n = wid & 1;
  const int half   = lane >> 4;
  const int nloc   = lane & 15;
  const int bb = blockIdx.x;
  const int cb = blockIdx.y;

  v8f acc[3][2];
  #pragma unroll
  for (int w = 0; w < 3; ++w)
    #pragma unroll
    for (int i = 0; i < 2; ++i)
      acc[w][i] = (v8f)(0.0f);

  stage_tiles(xpad, wbf, smem[0], bb, cb, 0, tid);

  for (int ki = 0; ki < INDIM / KSTEP; ++ki) {
    unsigned char* cur = smem[ki & 1];
    if (ki + 1 < INDIM / KSTEP) {
      stage_tiles(xpad, wbf, smem[(ki + 1) & 1], bb, cb, (ki + 1) * KSTEP, tid);
      wait_async<4>();           // the 4 older copies (cur buffer) have landed
    } else {
      wait_async<0>();
    }
    __syncthreads();

    const vu4* pa = (const vu4*)(cur + (wave_m * 16 + nloc) * ROWB);
    Frag a;
    a.q[0] = pa[half];
    a.q[1] = pa[half + 2];

    #pragma unroll
    for (int w = 0; w < 3; ++w) {
      #pragma unroll
      for (int i = 0; i < 2; ++i) {
        const int n = (wave_n * 2 + i) * 16 + nloc;
        const vu4* pb =
            (const vu4*)(cur + (NB * TPAD + w * NC + n) * ROWB);
        Frag bq;
        bq.q[0] = pb[half];
        bq.q[1] = pb[half + 2];
        acc[w][i] = __builtin_amdgcn_wmma_f32_16x16x32_bf16(
            false, a.v, false, bq.v, (short)0, acc[w][i], false, false);
      }
    }
    __syncthreads();             // reads done before this buffer is restaged
  }

  lif_scan_store(acc, bb * NB + wave_m, cb * NC + wave_n * 32, half, nloc,
                 b1, b2, b3, Wl[0], Wl[1], Wl[2], bl[0], out);
}

// ---- fallback: f32 inputs, in-kernel conversion (used if ws too small) ----
__global__ __launch_bounds__(256)
void lif_hh_fused_f32_kernel(const float* __restrict__ x,
                             const float* __restrict__ W1, const float* __restrict__ b1,
                             const float* __restrict__ W2, const float* __restrict__ b2,
                             const float* __restrict__ W3, const float* __restrict__ b3,
                             const float* __restrict__ Wl, const float* __restrict__ bl,
                             float* __restrict__ out)
{
  __shared__ __align__(16) unsigned char smem[TILEB];

  const int tid    = threadIdx.x;
  const int lane   = tid & 31;
  const int wid    = tid >> 5;
  const int wave_m = wid >> 1;
  const int wave_n = wid & 1;
  const int half   = lane >> 4;
  const int nloc   = lane & 15;
  const int bb = blockIdx.x;
  const int cb = blockIdx.y;

  unsigned char* lA = smem;
  unsigned char* lB = smem + NB * TPAD * ROWB;
  const float* Wmat[3] = {W1, W2, W3};

  v8f acc[3][2];
  #pragma unroll
  for (int w = 0; w < 3; ++w)
    #pragma unroll
    for (int i = 0; i < 2; ++i)
      acc[w][i] = (v8f)(0.0f);

  for (int k0 = 0; k0 < INDIM; k0 += KSTEP) {
    for (int s = tid; s < NB * TPAD * 8; s += 256) {
      const int row = s >> 3, part = s & 7;
      const int t = row & 15;
      const int b = bb * NB + (row >> 4);
      vf4 v = (vf4)(0.0f);
      if (t < TSTEP)
        v = *(const vf4*)(x + (size_t)(b * TSTEP + t) * INDIM + k0 + part * 4);
      vu2 pk;
      pk.x = pack2_bf16(v.x, v.y);
      pk.y = pack2_bf16(v.z, v.w);
      *(vu2*)(lA + row * ROWB + part * 8) = pk;
    }
    for (int s = tid; s < 3 * NC * 8; s += 256) {
      const int row = s >> 3, part = s & 7;
      const int w = row >> 6, n = row & 63;
      vf4 v = *(const vf4*)(Wmat[w] + (size_t)(cb * NC + n) * INDIM + k0 + part * 4);
      vu2 pk;
      pk.x = pack2_bf16(v.x, v.y);
      pk.y = pack2_bf16(v.z, v.w);
      *(vu2*)(lB + row * ROWB + part * 8) = pk;
    }
    __syncthreads();

    const vu4* pa = (const vu4*)(lA + (wave_m * 16 + nloc) * ROWB);
    Frag a;
    a.q[0] = pa[half];
    a.q[1] = pa[half + 2];

    #pragma unroll
    for (int w = 0; w < 3; ++w) {
      #pragma unroll
      for (int i = 0; i < 2; ++i) {
        const int n = (wave_n * 2 + i) * 16 + nloc;
        const vu4* pb = (const vu4*)(lB + (w * NC + n) * ROWB);
        Frag bq;
        bq.q[0] = pb[half];
        bq.q[1] = pb[half + 2];
        acc[w][i] = __builtin_amdgcn_wmma_f32_16x16x32_bf16(
            false, a.v, false, bq.v, (short)0, acc[w][i], false, false);
      }
    }
    __syncthreads();
  }

  lif_scan_store(acc, bb * NB + wave_m, cb * NC + wave_n * 32, half, nloc,
                 b1, b2, b3, Wl[0], Wl[1], Wl[2], bl[0], out);
}

extern "C" void kernel_launch(void* const* d_in, const int* in_sizes, int n_in,
                              void* d_out, int out_size, void* d_ws, size_t ws_size,
                              hipStream_t stream) {
  (void)in_sizes; (void)n_in; (void)out_size;
  const float* x  = (const float*)d_in[0];
  const float* W1 = (const float*)d_in[1];
  const float* b1 = (const float*)d_in[2];
  const float* W2 = (const float*)d_in[3];
  const float* b2 = (const float*)d_in[4];
  const float* W3 = (const float*)d_in[5];
  const float* b3 = (const float*)d_in[6];
  const float* Wl = (const float*)d_in[7];
  const float* bl = (const float*)d_in[8];
  float* out = (float*)d_out;

  dim3 grid(BATCH / NB, CDIM / NC);   // 512 x 16 workgroups

  if (ws_size >= WS_NEED) {
    __bf16* xpad = (__bf16*)d_ws;
    __bf16* wbf  = (__bf16*)((char*)d_ws + 2 * XPAD_ELEMS);

    cvt_x_kernel<<<(unsigned)(XPAD_ELEMS / 4 / 256), 256, 0, stream>>>(x, xpad);
    cvt_w_kernel<<<(unsigned)(CDIM * INDIM / 4 / 256), 256, 0, stream>>>(
        W1, wbf + 0 * (size_t)CDIM * INDIM);
    cvt_w_kernel<<<(unsigned)(CDIM * INDIM / 4 / 256), 256, 0, stream>>>(
        W2, wbf + 1 * (size_t)CDIM * INDIM);
    cvt_w_kernel<<<(unsigned)(CDIM * INDIM / 4 / 256), 256, 0, stream>>>(
        W3, wbf + 2 * (size_t)CDIM * INDIM);

    lif_hh_fused_bf16_kernel<<<grid, 256, 0, stream>>>(
        xpad, wbf, b1, b2, b3, Wl, bl, out);
  } else {
    lif_hh_fused_f32_kernel<<<grid, 256, 0, stream>>>(
        x, W1, b1, W2, b2, W3, b3, Wl, bl, out);
  }
}
